// Net_10239202034102
// MI455X (gfx1250) — matrix operand order repaired and checked by reading, not compiled, not run on hardware
//
#include <hip/hip_runtime.h>
#include <hip/hip_bf16.h>

// ---------------------------------------------------------------------------
// ResCorr stack, B=32768, L=6, I=512, H=256, O=512.
// s0,s1 are re-zeroed every layer -> each concat-GEMM is a sum over only the
// live 256-wide K segments.  bf16 WMMA (v_wmma_f32_16x16x32_bf16), f32 accum.
//
// This round: weight-traffic roofline fix.
//  * 32 rows per wave (2 M-tiles) -> every B-fragment feeds 2 WMMAs, halving
//    the dominant L2 weight stream (~22GB -> ~11GB per call).
//  * fused dual-GEMM passes for the tanh/sigmoid gate pairs (W0|W1, W2|W3):
//    one A sweep, two accumulator sets -> no f32 tmp buffer at all.
//  * 4 rotating bf16 LDS buffers (64KB total) ping-pong the states; no copy
//    loops, no in-place GEMM hazards, single-wave workgroups (no barriers).
// ---------------------------------------------------------------------------

typedef __attribute__((ext_vector_type(16))) __bf16 v16bf;
typedef __attribute__((ext_vector_type(8)))  float  v8f;

union FragU { v16bf v; uint4 u[2]; };

#define DEVINL __device__ __forceinline__

DEVINL unsigned short f2bf(float f) {
    union { float f; unsigned u; } c; c.f = f;
    unsigned r = c.u + 0x7fffu + ((c.u >> 16) & 1u);   // RNE
    return (unsigned short)(r >> 16);
}
DEVINL float bf2f(unsigned short s) {
    union { unsigned u; float f; } c; c.u = ((unsigned)s) << 16;
    return c.f;
}
DEVINL float sigmoid_f(float x) { return 1.0f / (1.0f + __expf(-x)); }
DEVINL float tanh_f(float x) {
    float ax = fabsf(x);
    float e  = __expf(-2.0f * ax);
    float t  = (1.0f - e) / (1.0f + e);
    return x >= 0.0f ? t : -t;
}

// ---- sizes -----------------------------------------------------------------
static constexpr int Hn = 256, In = 512, On = 512, Ln = 6;
static constexpr int MROWS = 32;       // rows per wave (2 M-tiles)

// workspace layout (bf16/ushort elements)
static constexpr long WE_OFF = 0;                                // [256][512]
static constexpr long WE_SZ  = (long)Hn * In;
static constexpr long WK_SZ  = (long)Ln * Hn * (3 * Hn);         // [6][256][768]
static constexpr long WK_OFF = WE_OFF + WE_SZ;                   // W0..W5
static constexpr long W6_OFF = WK_OFF + 6 * WK_SZ;               // [6][256][512]
static constexpr long W6_SZ  = (long)Ln * Hn * (2 * Hn);
static constexpr long WO_OFF = W6_OFF + W6_SZ;                   // [512][256]

// ---- fragment loads --------------------------------------------------------
// A (16x32 bf16) from LDS row-major [row][256]:
//   lane<16 : M=lane,    K = kk+0..7  and kk+16..23
//   lane>=16: M=lane-16, K = kk+8..15 and kk+24..31
DEVINL v16bf load_a(const unsigned short* seg, int kk, int lane) {
    const unsigned short* p = seg + (lane & 15) * Hn + kk + (lane >> 4) * 8;
    FragU f;
    f.u[0] = *(const uint4*)(p);
    f.u[1] = *(const uint4*)(p + 16);
    return f.v;
}
// B (32x16 bf16) from global Wt[n][k] (k contiguous):
//   lane<16 : N=lane,    K = kb+0..7 and kb+16..23 ;  lane>=16: N=lane-16, K = kb+8..15 and kb+24..31
DEVINL v16bf load_b(const unsigned short* w, int Ktot, int ntile, int kb, int lane) {
    const unsigned short* p = w + (long)(ntile * 16 + (lane & 15)) * Ktot + kb + (lane >> 4) * 8;
    FragU f;
    f.u[0] = *(const uint4*)(p);
    f.u[1] = *(const uint4*)(p + 16);
    return f.v;
}

DEVINL v16bf wmma_bf16(v16bf a, v16bf b, v8f& acc);  // fwd decl (unused)

// ---- single-weight GEMM: 32 rows x (ntiles*16) cols, K = 256*nseg ---------
template <int NTG, typename F>
DEVINL void gemmN(const unsigned short* const* segs, int nseg,
                  const unsigned short* w, int Ktot, const float* bias,
                  int ntiles, int lane, F&& consume) {
    for (int nb = 0; nb < ntiles; nb += NTG) {
        v8f acc[2][NTG];
        #pragma unroll
        for (int nt = 0; nt < NTG; ++nt) {
            float bv = bias[(nb + nt) * 16 + (lane & 15)];
            #pragma unroll
            for (int i = 0; i < 8; ++i) { acc[0][nt][i] = bv; acc[1][nt][i] = bv; }
        }
        for (int s = 0; s < nseg; ++s) {
            #pragma unroll 2
            for (int kk = 0; kk < 256; kk += 32) {
                v16bf a0 = load_a(segs[s], kk, lane);
                v16bf a1 = load_a(segs[s] + 16 * Hn, kk, lane);
                #pragma unroll
                for (int nt = 0; nt < NTG; ++nt) {
                    v16bf b = load_b(w, Ktot, nb + nt, s * 256 + kk, lane);
                    acc[0][nt] = __builtin_amdgcn_wmma_f32_16x16x32_bf16(
                        false, a0, false, b, (short)0, acc[0][nt], false, false);
                    acc[1][nt] = __builtin_amdgcn_wmma_f32_16x16x32_bf16(
                        false, a1, false, b, (short)0, acc[1][nt], false, false);
                }
            }
        }
        #pragma unroll
        for (int nt = 0; nt < NTG; ++nt)
            #pragma unroll
            for (int mt = 0; mt < 2; ++mt) consume(nb + nt, mt, acc[mt][nt]);
    }
}

// ---- fused dual-weight GEMM (shared A sweep, two accumulator sets) ---------
template <int NTG, typename F>
DEVINL void gemmDual(const unsigned short* const* segs, int nseg,
                     const unsigned short* wT, const unsigned short* wG, int Ktot,
                     const float* bT, const float* bG,
                     int ntiles, int lane, F&& consume) {
    for (int nb = 0; nb < ntiles; nb += NTG) {
        v8f accT[2][NTG], accG[2][NTG];
        #pragma unroll
        for (int nt = 0; nt < NTG; ++nt) {
            float bvT = bT[(nb + nt) * 16 + (lane & 15)];
            float bvG = bG[(nb + nt) * 16 + (lane & 15)];
            #pragma unroll
            for (int i = 0; i < 8; ++i) {
                accT[0][nt][i] = bvT; accT[1][nt][i] = bvT;
                accG[0][nt][i] = bvG; accG[1][nt][i] = bvG;
            }
        }
        for (int s = 0; s < nseg; ++s) {
            #pragma unroll 2
            for (int kk = 0; kk < 256; kk += 32) {
                v16bf a0 = load_a(segs[s], kk, lane);
                v16bf a1 = load_a(segs[s] + 16 * Hn, kk, lane);
                #pragma unroll
                for (int nt = 0; nt < NTG; ++nt) {
                    v16bf bt = load_b(wT, Ktot, nb + nt, s * 256 + kk, lane);
                    accT[0][nt] = __builtin_amdgcn_wmma_f32_16x16x32_bf16(
                        false, a0, false, bt, (short)0, accT[0][nt], false, false);
                    accT[1][nt] = __builtin_amdgcn_wmma_f32_16x16x32_bf16(
                        false, a1, false, bt, (short)0, accT[1][nt], false, false);
                    v16bf bg = load_b(wG, Ktot, nb + nt, s * 256 + kk, lane);
                    accG[0][nt] = __builtin_amdgcn_wmma_f32_16x16x32_bf16(
                        false, a0, false, bg, (short)0, accG[0][nt], false, false);
                    accG[1][nt] = __builtin_amdgcn_wmma_f32_16x16x32_bf16(
                        false, a1, false, bg, (short)0, accG[1][nt], false, false);
                }
            }
        }
        #pragma unroll
        for (int nt = 0; nt < NTG; ++nt)
            #pragma unroll
            for (int mt = 0; mt < 2; ++mt)
                consume(nb + nt, mt, accT[mt][nt], accG[mt][nt]);
    }
}

// ---- weight transpose/convert: src [Lb][K][N] f32 -> dst [Lb][N][K] bf16 ---
__global__ void wprep_kernel(const float* __restrict__ src, unsigned short* __restrict__ dst,
                             int Lb, int K, int N) {
    long total = (long)Lb * K * N;
    long stride = (long)gridDim.x * blockDim.x;
    for (long i = (long)blockIdx.x * blockDim.x + threadIdx.x; i < total; i += stride) {
        long kn = (long)K * N;
        int  l  = (int)(i / kn);
        long r  = i - (long)l * kn;
        int  k  = (int)(r / N);
        int  n  = (int)(r - (long)k * N);
        dst[((long)l * N + n) * K + k] = f2bf(src[i]);
    }
}

// ---- main fused network kernel: 1 wave / 32 rows / 64KB LDS ----------------
__global__ __launch_bounds__(32) void rescorr_kernel(
    const float* __restrict__ x,  const float* __restrict__ be,
    const float* __restrict__ b0, const float* __restrict__ b1,
    const float* __restrict__ b2, const float* __restrict__ b3,
    const float* __restrict__ b4, const float* __restrict__ b5,
    const float* __restrict__ b6, const float* __restrict__ bo,
    const unsigned short* __restrict__ ws, float* __restrict__ out) {

    // 4 rotating bf16 buffers, each [32][256] = 16KB -> 64KB total
    __shared__ unsigned short P0[MROWS * Hn], P1[MROWS * Hn],
                              P2[MROWS * Hn], P3[MROWS * Hn];

    const int lane = threadIdx.x & 31;
    const int col  = lane & 15;
    const int mb   = (lane >> 4) * 8;
    const long rowB = (long)blockIdx.x * MROWS;

    // ---- stage x tile: cols 0..255 -> P0, cols 256..511 -> P1 (f32->bf16) --
    {
        const float4* x4 = (const float4*)(x + rowB * In);
        for (int i = lane; i < MROWS * (In / 4); i += 32) {
            int row = i >> 7, c4 = i & 127;
            float4 v = x4[i];
            uint2 pk;
            pk.x = (unsigned)f2bf(v.x) | ((unsigned)f2bf(v.y) << 16);
            pk.y = (unsigned)f2bf(v.z) | ((unsigned)f2bf(v.w) << 16);
            unsigned short* dst = (c4 < 64) ? P0 : P1;
            *(uint2*)(dst + row * Hn + (c4 & 63) * 4) = pk;
        }
    }

    // ---- h = x @ We + be  -> P2 --------------------------------------------
    {
        const unsigned short* segs[2] = { P0, P1 };
        gemmN<4>(segs, 2, ws + WE_OFF, In, be, Hn / 16, lane,
            [&](int nt, int mt, v8f a) {
                #pragma unroll
                for (int r = 0; r < 8; ++r)
                    P2[(mt * 16 + mb + r) * Hn + nt * 16 + col] = f2bf(a[r]);
            });
    }

    unsigned short *hp = P2, *fA = P0, *fB = P1, *fC = P3;

    for (int l = 0; l < Ln; ++l) {
        const unsigned short* w0 = ws + WK_OFF + 0 * WK_SZ + (long)l * Hn * 768;
        const unsigned short* w1 = ws + WK_OFF + 1 * WK_SZ + (long)l * Hn * 768;
        const unsigned short* w2 = ws + WK_OFF + 2 * WK_SZ + (long)l * Hn * 768;
        const unsigned short* w3 = ws + WK_OFF + 3 * WK_SZ + (long)l * Hn * 768;
        const unsigned short* w4 = ws + WK_OFF + 4 * WK_SZ + (long)l * Hn * 768;
        const unsigned short* w5 = ws + WK_OFF + 5 * WK_SZ + (long)l * Hn * 768;
        const unsigned short* w6 = ws + W6_OFF + (long)l * Hn * 512;
        const float* pb0 = b0 + l * Hn; const float* pb1 = b1 + l * Hn;
        const float* pb2 = b2 + l * Hn; const float* pb3 = b3 + l * Hn;
        const float* pb4 = b4 + l * Hn; const float* pb5 = b5 + l * Hn;
        const float* pb6 = b6 + l * Hn;

        // stream upcoming weights toward caches (gfx1250 global_prefetch)
        __builtin_prefetch(w4 + lane * 64, 0, 1);
        __builtin_prefetch(w2 + lane * 64, 0, 1);

        const unsigned short* segH[1] = { hp };

        // AB: ns0 = sigmoid(h@W1+b1) * tanh(h@W0+b0)   -> fA   (s0=s1=0)
        gemmDual<2>(segH, 1, w0, w1, 768, pb0, pb1, Hn / 16, lane,
            [&](int nt, int mt, v8f t, v8f g) {
                #pragma unroll
                for (int r = 0; r < 8; ++r) {
                    int idx = (mt * 16 + mb + r) * Hn + nt * 16 + col;
                    fA[idx] = f2bf(sigmoid_f(g[r]) * tanh_f(t[r]));
                }
            });

        // C: ns0 *= sigmoid([h,ns0]@W4+b4)             fA -> fB
        {
            const unsigned short* segs[2] = { hp, fA };
            gemmN<4>(segs, 2, w4, 768, pb4, Hn / 16, lane,
                [&](int nt, int mt, v8f a) {
                    #pragma unroll
                    for (int r = 0; r < 8; ++r) {
                        int idx = (mt * 16 + mb + r) * Hn + nt * 16 + col;
                        fB[idx] = f2bf(sigmoid_f(a[r]) * bf2f(fA[idx]));
                    }
                });
        }

        // DE: ns1 = sigmoid([h,ns0]@W3+b3) * tanh([h,ns0]@W2+b2)  -> fC
        {
            const unsigned short* segs[2] = { hp, fB };
            gemmDual<2>(segs, 2, w2, w3, 768, pb2, pb3, Hn / 16, lane,
                [&](int nt, int mt, v8f t, v8f g) {
                    #pragma unroll
                    for (int r = 0; r < 8; ++r) {
                        int idx = (mt * 16 + mb + r) * Hn + nt * 16 + col;
                        fC[idx] = f2bf(sigmoid_f(g[r]) * tanh_f(t[r]));
                    }
                });
        }

        // F: ns0 = ns0 + sigmoid([h,ns0,ns1]@W5+b5)*ns1           -> fA
        {
            const unsigned short* segs[3] = { hp, fB, fC };
            gemmN<4>(segs, 3, w5, 768, pb5, Hn / 16, lane,
                [&](int nt, int mt, v8f a) {
                    #pragma unroll
                    for (int r = 0; r < 8; ++r) {
                        int idx = (mt * 16 + mb + r) * Hn + nt * 16 + col;
                        fA[idx] = f2bf(bf2f(fB[idx]) + sigmoid_f(a[r]) * bf2f(fC[idx]));
                    }
                });
        }

        // G: h' = silu([ns0,h]@W6+b6)                              -> fB
        {
            const unsigned short* segs[2] = { fA, hp };
            gemmN<4>(segs, 2, w6, 512, pb6, Hn / 16, lane,
                [&](int nt, int mt, v8f a) {
                    #pragma unroll
                    for (int r = 0; r < 8; ++r) {
                        int idx = (mt * 16 + mb + r) * Hn + nt * 16 + col;
                        float v = a[r];
                        fB[idx] = f2bf(v * sigmoid_f(v));
                    }
                });
        }

        // rotate: new h = fB; old hp/fA/fC become free
        unsigned short* oldh = hp;
        hp = fB; fB = fA; fA = oldh;   // fC stays
    }

    // ---- out = h @ Wo + bo (direct f32 stores) ------------------------------
    {
        const unsigned short* segH[1] = { hp };
        gemmN<4>(segH, 1, ws + WO_OFF, Hn, bo, On / 16, lane,
            [&](int nt, int mt, v8f a) {
                #pragma unroll
                for (int r = 0; r < 8; ++r)
                    out[(rowB + mt * 16 + mb + r) * On + nt * 16 + col] = a[r];
            });
    }
}

// ---------------------------------------------------------------------------
extern "C" void kernel_launch(void* const* d_in, const int* in_sizes, int n_in,
                              void* d_out, int out_size, void* d_ws, size_t ws_size,
                              hipStream_t stream) {
    const float* x  = (const float*)d_in[0];
    const float* We = (const float*)d_in[1];
    const float* be = (const float*)d_in[2];
    const float* W0 = (const float*)d_in[3];  const float* b0 = (const float*)d_in[4];
    const float* W1 = (const float*)d_in[5];  const float* b1 = (const float*)d_in[6];
    const float* W2 = (const float*)d_in[7];  const float* b2 = (const float*)d_in[8];
    const float* W3 = (const float*)d_in[9];  const float* b3 = (const float*)d_in[10];
    const float* W4 = (const float*)d_in[11]; const float* b4 = (const float*)d_in[12];
    const float* W5 = (const float*)d_in[13]; const float* b5 = (const float*)d_in[14];
    const float* W6 = (const float*)d_in[15]; const float* b6 = (const float*)d_in[16];
    const float* Wo = (const float*)d_in[17]; const float* bo = (const float*)d_in[18];
    float* out = (float*)d_out;
    unsigned short* ws = (unsigned short*)d_ws;   // needs ~15.6 MB scratch

    const int B = in_sizes[0] / In;

    // ---- prep: transpose+convert all weights to bf16 Wt[n][k] --------------
    auto prep = [&](const float* src, long dstOff, int Lb, int K, int N) {
        long total = (long)Lb * K * N;
        int blocks = (int)((total + 255) / 256);
        if (blocks > 2048) blocks = 2048;
        wprep_kernel<<<blocks, 256, 0, stream>>>(src, ws + dstOff, Lb, K, N);
    };
    prep(We, WE_OFF, 1, In, Hn);
    prep(W0, WK_OFF + 0 * WK_SZ, Ln, 3 * Hn, Hn);
    prep(W1, WK_OFF + 1 * WK_SZ, Ln, 3 * Hn, Hn);
    prep(W2, WK_OFF + 2 * WK_SZ, Ln, 3 * Hn, Hn);
    prep(W3, WK_OFF + 3 * WK_SZ, Ln, 3 * Hn, Hn);
    prep(W4, WK_OFF + 4 * WK_SZ, Ln, 3 * Hn, Hn);
    prep(W5, WK_OFF + 5 * WK_SZ, Ln, 3 * Hn, Hn);
    prep(W6, W6_OFF, Ln, 2 * Hn, Hn);
    prep(Wo, WO_OFF, 1, Hn, On);

    // ---- fused network: 1 wave per block, 32 rows, 64KB static LDS ----------
    rescorr_kernel<<<dim3(B / MROWS), dim3(32), 0, stream>>>(
        x, be, b0, b1, b2, b3, b4, b5, b6, bo, ws, out);
}
